// MambaBlock_16853451670080
// MI455X (gfx1250) — compile-verified, matrix operand order checked by reading, not run on hardware
//
#include <hip/hip_runtime.h>
#include <hip/hip_bf16.h>

// ---------------------------------------------------------------------------
// Mamba block for MI455X (gfx1250, wave32).
// GEMMs: WMMA bf16 pipe (v_wmma_f32_16x16x32_bf16, f32 accum) with
// double-buffered async global->LDS staging (global_load_async_to_lds_b128,
// ASYNCcnt) shared by the 8 waves of each workgroup. Weights converted +
// transposed to bf16 once per launch so all fragment reads are contiguous.
// Selective scan: 1 thread per (batch, channel), state in registers.
// ---------------------------------------------------------------------------

#define BATCH   4
#define SEQ     2048
#define DMODEL  1024
#define DINNER  2048
#define NSTATE  16
#define DCONV   4
#define NTOK    (BATCH * SEQ)          // 8192 tokens

#define BLK_M   64
#define BLK_N   64
#define BLK_K   32

typedef __attribute__((ext_vector_type(16))) __bf16 v16bf;
typedef __attribute__((ext_vector_type(8)))  float  v8f;

static_assert(sizeof(__bf16) == 2, "bf16 must be 2 bytes");

__device__ __forceinline__ float silu_f(float v) {
    return v / (1.0f + __expf(-v));
}

__device__ __forceinline__ unsigned lds_off(const void* p) {
    // LDS aperture: generic address low 32 bits are the LDS byte address.
    return (unsigned)(uintptr_t)p;
}

// ---------------------------------------------------------------------------
// Convert fp32 weight [K,N] (row-major) to bf16 transposed copy [N,K].
// ---------------------------------------------------------------------------
__global__ void cvt_transpose_kernel(const float* __restrict__ W,
                                     __bf16* __restrict__ WT,
                                     int K, int N) {
    size_t i = (size_t)blockIdx.x * blockDim.x + threadIdx.x;
    size_t total = (size_t)K * N;
    if (i >= total) return;
    int k = (int)(i / N);
    int n = (int)(i % N);
    WT[(size_t)n * K + k] = (__bf16)W[i];
}

// ---------------------------------------------------------------------------
// LayerNorm over DMODEL, one block per token, output bf16.
// ---------------------------------------------------------------------------
__global__ void layernorm_kernel(const float* __restrict__ x,
                                 const float* __restrict__ w,
                                 const float* __restrict__ b,
                                 __bf16* __restrict__ hbf) {
    const int D = DMODEL;
    const int token = blockIdx.x;
    const float* xp = x + (size_t)token * D;

    float s = 0.f, ss = 0.f;
    for (int i = threadIdx.x; i < D; i += blockDim.x) {
        float v = xp[i];
        s += v; ss += v * v;
    }
    for (int off = 16; off > 0; off >>= 1) {
        s  += __shfl_down(s,  off, 32);
        ss += __shfl_down(ss, off, 32);
    }
    __shared__ float sred[8], ssred[8];
    const int wid = threadIdx.x >> 5;
    if ((threadIdx.x & 31) == 0) { sred[wid] = s; ssred[wid] = ss; }
    __syncthreads();
    if (threadIdx.x == 0) {
        float ts = 0.f, tss = 0.f;
        const int nw = blockDim.x >> 5;
        for (int i = 0; i < nw; ++i) { ts += sred[i]; tss += ssred[i]; }
        sred[0] = ts; ssred[0] = tss;
    }
    __syncthreads();
    const float mean = sred[0] / (float)D;
    const float var  = ssred[0] / (float)D - mean * mean;
    const float rstd = rsqrtf(var + 1e-5f);
    for (int i = threadIdx.x; i < D; i += blockDim.x) {
        float v = (xp[i] - mean) * rstd * w[i] + b[i];
        hbf[(size_t)token * D + i] = (__bf16)v;
    }
}

// ---------------------------------------------------------------------------
// WMMA GEMM with double-buffered async global->LDS staging.
//   out[M,N] = A[M,K](bf16) x BT[N,K]^T(bf16) + bias[N]  (+ epilogue)
// Block: 256 thr = 8 waves; block tile 64x64, K-step 32.
// Wave w: M rows [(w>>2)*32, +32), N cols [(w&3)*16, +16) => 2 WMMA tiles
// sharing one B fragment.
// Fragment mapping (contiguous k order, same for A and B), lane L, r=L&15, h=L>>4:
//   A frag elem i = sA[(mloc + r)*32 + 16h + i]   (32B contiguous LDS read)
//   B frag elem i = sB[(nloc + r)*32 + 16h + i]
//   D (f32, 8 VGPRs): vgpr v -> row m + 8h + v, col n + r.
// EPI: 0 = bias only, 1 = softplus, 2 = +bias +residual.
// ---------------------------------------------------------------------------
template <int EPI>
__global__ __launch_bounds__(256) void gemm_wmma_kernel(
        const __bf16* __restrict__ A,
        const __bf16* __restrict__ BT,
        const float*  __restrict__ bias,
        const float*  __restrict__ resid,
        float* __restrict__ out,
        int M, int N, int K) {
    __shared__ __attribute__((aligned(32))) __bf16 sA[2][BLK_M * BLK_K];
    __shared__ __attribute__((aligned(32))) __bf16 sB[2][BLK_N * BLK_K];

    const int tilesN = N / BLK_N;
    const int m0 = (int)(blockIdx.x / tilesN) * BLK_M;
    const int n0 = (int)(blockIdx.x % tilesN) * BLK_N;

    const int tid = threadIdx.x;
    // Stage-copy mapping: thread t moves 16B of A and 16B of B per stage.
    // Tile = 64 rows x 32 bf16 (64B/row): row = t>>2, 16B segment = t&3.
    const int cpRow = tid >> 2;
    const int cpSeg = tid & 3;
    const __bf16* gA = A  + (size_t)(m0 + cpRow) * K + cpSeg * 8;
    const __bf16* gB = BT + (size_t)(n0 + cpRow) * K + cpSeg * 8;
    const int cpIdx = cpRow * BLK_K + cpSeg * 8;

    // wave tile assignment
    const int wave = tid >> 5;
    const int lane = tid & 31;
    const int r    = lane & 15;
    const int half = lane >> 4;
    const int wm   = (wave >> 2) * 32;     // 0 / 32
    const int wn   = (wave & 3) * 16;      // 0,16,32,48

    auto issue_stage = [&](int buf, int k0) {
        unsigned la = lds_off(&sA[buf][cpIdx]);
        unsigned lb = lds_off(&sB[buf][cpIdx]);
        const __bf16* pa = gA + k0;
        const __bf16* pb = gB + k0;
        // async DMA: 16B per lane, whole block covers the 4KB A + 4KB B tiles
        asm volatile("global_load_async_to_lds_b128 %0, %1, off\n\t"
                     "global_load_async_to_lds_b128 %2, %3, off"
                     :: "v"(la), "v"(pa), "v"(lb), "v"(pb)
                     : "memory");
    };

    v8f acc0 = {};
    v8f acc1 = {};

    issue_stage(0, 0);
    int buf = 0;
    for (int k0 = 0; k0 < K; k0 += BLK_K) {
        asm volatile("s_wait_asynccnt 0" ::: "memory");   // own stage landed
        __syncthreads();                                  // everyone's landed
        if (k0 + BLK_K < K)
            issue_stage(buf ^ 1, k0 + BLK_K);             // overlap next copy

        const v16bf a0 = *(const v16bf*)&sA[buf][(wm +      r) * BLK_K + half * 16];
        const v16bf a1 = *(const v16bf*)&sA[buf][(wm + 16 + r) * BLK_K + half * 16];
        const v16bf bb = *(const v16bf*)&sB[buf][(wn +      r) * BLK_K + half * 16];

        acc0 = __builtin_amdgcn_wmma_f32_16x16x32_bf16(
            false, a0, false, bb, (short)0, acc0, false, false);
        acc1 = __builtin_amdgcn_wmma_f32_16x16x32_bf16(
            false, a1, false, bb, (short)0, acc1, false, false);

        __syncthreads();                                  // done reading buf
        buf ^= 1;
    }

    const int   col = n0 + wn + r;
    const float bc  = bias[col];
#pragma unroll
    for (int t = 0; t < 2; ++t) {
        const v8f& acc = t ? acc1 : acc0;
        const int row0 = m0 + wm + t * 16 + half * 8;
#pragma unroll
        for (int v = 0; v < 8; ++v) {
            float val = acc[v] + bc;
            if (EPI == 1) {                       // softplus
                val = (val > 20.f) ? val : log1pf(__expf(val));
            } else if (EPI == 2) {                // + residual
                val += resid[(size_t)(row0 + v) * N + col];
            }
            out[(size_t)(row0 + v) * N + col] = val;
        }
    }
}

// ---------------------------------------------------------------------------
// xin = silu(xr[:, :DINNER]); depthwise causal conv(4) + bias; silu.
// ---------------------------------------------------------------------------
__global__ void conv_silu_kernel(const float* __restrict__ xr,    // [NTOK, 2*DINNER]
                                 const float* __restrict__ cw,    // [DINNER,1,DCONV]
                                 const float* __restrict__ cb,    // [DINNER]
                                 float* __restrict__ xs,
                                 __bf16* __restrict__ xsbf) {
    size_t i = (size_t)blockIdx.x * blockDim.x + threadIdx.x;
    if (i >= (size_t)NTOK * DINNER) return;
    const int d     = (int)(i % DINNER);
    const size_t bl = i / DINNER;
    const int l     = (int)(bl % SEQ);

    float acc = cb[d];
#pragma unroll
    for (int j = 0; j < DCONV; ++j) {
        const int lp = l - (DCONV - 1) + j;
        if (lp >= 0) {
            const size_t row = bl - (size_t)(DCONV - 1 - j);
            float v = xr[row * (2 * DINNER) + d];
            acc += cw[d * DCONV + j] * silu_f(v);
        }
    }
    const float o = silu_f(acc);
    xs[i]   = o;
    xsbf[i] = (__bf16)o;
}

// ---------------------------------------------------------------------------
// x_proj: BC[token, 0:32] = xs[token,:] @ xproj_w[2048,32] + xproj_b.
// ---------------------------------------------------------------------------
__global__ void xproj_kernel(const float* __restrict__ xs,
                             const float* __restrict__ w,
                             const float* __restrict__ b,
                             float* __restrict__ BC) {
    const int NCOL = 2 * NSTATE;                 // 32
    const int token = blockIdx.x;
    const int col = threadIdx.x & (NCOL - 1);
    const int seg = threadIdx.x / NCOL;          // 0..7
    const float* xp = xs + (size_t)token * DINNER;

    float acc = 0.f;
    for (int k = seg; k < DINNER; k += 8)
        acc += xp[k] * w[(size_t)k * NCOL + col];

    __shared__ float red[8][NCOL];
    red[seg][col] = acc;
    __syncthreads();
    if (seg == 0) {
        float t = acc;
#pragma unroll
        for (int s2 = 1; s2 < 8; ++s2) t += red[s2][col];
        BC[(size_t)token * NCOL + col] = t + b[col];
    }
}

// ---------------------------------------------------------------------------
// Selective scan + D skip + SiLU(res) gating.
// ---------------------------------------------------------------------------
__global__ void scan_kernel(const float* __restrict__ xs,      // [NTOK, DINNER]
                            const float* __restrict__ delta,   // [NTOK, DINNER]
                            const float* __restrict__ Alog,    // [DINNER, NSTATE]
                            const float* __restrict__ Dp,      // [DINNER]
                            const float* __restrict__ BC,      // [NTOK, 32]
                            const float* __restrict__ xr,      // res at col DINNER
                            __bf16* __restrict__ ybf) {
    const int blocksPerB = DINNER / 256;                       // 8
    const int bidx = blockIdx.x / blocksPerB;
    const int d    = (blockIdx.x % blocksPerB) * 256 + threadIdx.x;

    float a[NSTATE], h[NSTATE];
#pragma unroll
    for (int n = 0; n < NSTATE; ++n) {
        a[n] = -__expf(Alog[(size_t)d * NSTATE + n]);
        h[n] = 0.f;
    }
    const float Dd = Dp[d];

    __shared__ float sBC[2 * NSTATE];
    const size_t base = (size_t)bidx * SEQ;

    for (int l = 0; l < SEQ; ++l) {
        const size_t tok = base + l;
        if (threadIdx.x < 2 * NSTATE)
            sBC[threadIdx.x] = BC[tok * (2 * NSTATE) + threadIdx.x];
        __syncthreads();

        const float dt  = delta[tok * DINNER + d];
        const float xv  = xs[tok * DINNER + d];
        const float dtx = dt * xv;
        float y = 0.f;
#pragma unroll
        for (int n = 0; n < NSTATE; ++n) {
            h[n] = h[n] * __expf(dt * a[n]) + dtx * sBC[n];
            y += h[n] * sBC[NSTATE + n];
        }
        y += Dd * xv;
        const float res = xr[tok * (2 * DINNER) + DINNER + d];
        y *= silu_f(res);
        ybf[tok * DINNER + d] = (__bf16)y;
        __syncthreads();
    }
}

// ---------------------------------------------------------------------------
// Host launcher
// ---------------------------------------------------------------------------
extern "C" void kernel_launch(void* const* d_in, const int* in_sizes, int n_in,
                              void* d_out, int out_size, void* d_ws, size_t ws_size,
                              hipStream_t stream) {
    const float* x       = (const float*)d_in[0];
    const float* norm_w  = (const float*)d_in[1];
    const float* norm_b  = (const float*)d_in[2];
    const float* in_w    = (const float*)d_in[3];
    const float* in_b    = (const float*)d_in[4];
    const float* conv_w  = (const float*)d_in[5];
    const float* conv_b  = (const float*)d_in[6];
    const float* xproj_w = (const float*)d_in[7];
    const float* xproj_b = (const float*)d_in[8];
    const float* dt_w    = (const float*)d_in[9];
    const float* dt_b    = (const float*)d_in[10];
    const float* A_log   = (const float*)d_in[11];
    const float* D_param = (const float*)d_in[12];
    const float* out_w   = (const float*)d_in[13];
    const float* out_b   = (const float*)d_in[14];
    float* out = (float*)d_out;

    char* ws = (char*)d_ws;
    size_t off = 0;
    auto carve = [&](size_t bytes) -> void* {
        void* p = ws + off;
        off = (off + bytes + 255) & ~(size_t)255;
        return p;
    };
    __bf16* hbf    = (__bf16*)carve((size_t)NTOK * DMODEL * 2);
    __bf16* wT_in  = (__bf16*)carve((size_t)(2 * DINNER) * DMODEL * 2);
    __bf16* wT_dt  = (__bf16*)carve((size_t)DINNER * DINNER * 2);
    __bf16* wT_out = (__bf16*)carve((size_t)DMODEL * DINNER * 2);
    float*  xr     = (float*) carve((size_t)NTOK * 2 * DINNER * 4);
    float*  xs     = (float*) carve((size_t)NTOK * DINNER * 4);
    __bf16* xsbf   = (__bf16*)carve((size_t)NTOK * DINNER * 2);
    float*  BC     = (float*) carve((size_t)NTOK * 2 * NSTATE * 4);
    float*  delta  = (float*) carve((size_t)NTOK * DINNER * 4);
    __bf16* ybf    = (__bf16*)carve((size_t)NTOK * DINNER * 2);
    (void)ws_size; (void)n_in; (void)in_sizes; (void)out_size;

    const int TPB = 256;

    // 1) weight convert + transpose -> bf16 [N,K]
    {
        size_t t0 = (size_t)DMODEL * (2 * DINNER);
        cvt_transpose_kernel<<<(unsigned)((t0 + TPB - 1) / TPB), TPB, 0, stream>>>(
            in_w, wT_in, DMODEL, 2 * DINNER);
        size_t t1 = (size_t)DINNER * DINNER;
        cvt_transpose_kernel<<<(unsigned)((t1 + TPB - 1) / TPB), TPB, 0, stream>>>(
            dt_w, wT_dt, DINNER, DINNER);
        size_t t2 = (size_t)DINNER * DMODEL;
        cvt_transpose_kernel<<<(unsigned)((t2 + TPB - 1) / TPB), TPB, 0, stream>>>(
            out_w, wT_out, DINNER, DMODEL);
    }

    // 2) LayerNorm -> bf16
    layernorm_kernel<<<NTOK, TPB, 0, stream>>>(x, norm_w, norm_b, hbf);

    // 3) in_proj GEMM: [8192,1024] x [1024,4096]
    gemm_wmma_kernel<0><<<(NTOK / BLK_M) * ((2 * DINNER) / BLK_N), TPB, 0, stream>>>(
        hbf, wT_in, in_b, nullptr, xr, NTOK, 2 * DINNER, DMODEL);

    // 4) silu + depthwise causal conv + silu
    {
        size_t tot = (size_t)NTOK * DINNER;
        conv_silu_kernel<<<(unsigned)((tot + TPB - 1) / TPB), TPB, 0, stream>>>(
            xr, conv_w, conv_b, xs, xsbf);
    }

    // 5) x_proj (B, C)
    xproj_kernel<<<NTOK, TPB, 0, stream>>>(xs, xproj_w, xproj_b, BC);

    // 6) dt_proj GEMM + softplus: [8192,2048] x [2048,2048]
    gemm_wmma_kernel<1><<<(NTOK / BLK_M) * (DINNER / BLK_N), TPB, 0, stream>>>(
        xsbf, wT_dt, dt_b, nullptr, delta, NTOK, DINNER, DINNER);

    // 7) selective scan + gating -> bf16 activation
    scan_kernel<<<BATCH * (DINNER / 256), TPB, 0, stream>>>(
        xs, delta, A_log, D_param, BC, xr, ybf);

    // 8) out_proj GEMM + bias + residual: [8192,2048] x [2048,1024]
    gemm_wmma_kernel<2><<<(NTOK / BLK_M) * (DMODEL / BLK_N), TPB, 0, stream>>>(
        ybf, wT_out, out_b, x, out, NTOK, DMODEL, DINNER);
}
